// DisentangledMHA_10033043604192
// MI455X (gfx1250) — matrix operand order repaired and checked by reading, not compiled, not run on hardware
//
#include <hip/hip_runtime.h>
#include <hip/hip_bf16.h>
#include <stdint.h>

// Problem constants (match reference)
#define B_   4
#define S_   512
#define DIM_ 1024
#define H_   16
#define HD_  64
static constexpr float SCORE_SCALE = 0.07216878364870323f; // (64*3)^-0.5

typedef __attribute__((ext_vector_type(16))) __bf16 v16bf;
typedef __attribute__((ext_vector_type(8)))  float  v8f;

union FragU { v16bf v; uint4 u[2]; };

// round-to-nearest-even fp32 -> bf16 (raw u16)
__device__ inline unsigned short f2bf(float f) {
    union { float f; unsigned u; } x; x.f = f;
    unsigned r = x.u + 0x7FFFu + ((x.u >> 16) & 1u);
    return (unsigned short)(r >> 16);
}

// A fragment 16x32 bf16 (M x K): lane L<16 -> row L, K = {0..7, 16..23};
// lane L>=16 -> row L-16, K = {8..15, 24..31}.  Two 16B contiguous loads.
__device__ inline v16bf load_afrag(const unsigned short* Abase, int lda,
                                   int row0, int kk, int lane) {
    const int r = row0 + (lane & 15);
    const int c = kk + ((lane & 16) ? 8 : 0);
    FragU f;
    f.u[0] = *reinterpret_cast<const uint4*>(Abase + (size_t)r * lda + c);
    f.u[1] = *reinterpret_cast<const uint4*>(Abase + (size_t)r * lda + c + 16);
    return f.v;
}

// B fragment 32x16 bf16 (K x N), supplied as B^T row-major (row = output col n,
// K contiguous): lane L<16 -> col L, K = {0..7, 16..23}; lane>=16 -> col L-16, K = {8..15, 24..31}.
__device__ inline v16bf load_bfrag(const unsigned short* Btbase, int ldb,
                                   int n0, int kk, int lane) {
    const int n = n0 + (lane & 15);
    const int c = kk + ((lane & 16) ? 8 : 0);
    FragU f;
    const uint4* p = reinterpret_cast<const uint4*>(Btbase + (size_t)n * ldb + c);
    f.u[0] = p[0];
    f.u[1] = p[2];
    return f.v;
}

__device__ inline v8f wmma_bf16(v16bf a, v16bf b, v8f c) {
    return __builtin_amdgcn_wmma_f32_16x16x32_bf16(
        /*neg_a=*/false, a, /*neg_b=*/false, b,
        /*c_mod=*/(short)0, c, /*reuse_a=*/false, /*reuse_b=*/false);
}

// ---------------------------------------------------------------- conversions
__global__ __launch_bounds__(256) void cvt_bf16_kernel(const float* __restrict__ in,
                                                       unsigned short* __restrict__ out,
                                                       int n) {
    int i = blockIdx.x * 256 + threadIdx.x;
    if (i < n) out[i] = f2bf(in[i]);
}

// Wt[n][k] = W[k][n], 1024x1024, fp32 -> bf16
__global__ __launch_bounds__(256) void transpose_cvt_kernel(const float* __restrict__ W,
                                                            unsigned short* __restrict__ Wt) {
    int i = blockIdx.x * 256 + threadIdx.x;   // output linear index
    int n = i >> 10;
    int k = i & 1023;
    Wt[i] = f2bf(W[k * 1024 + n]);
}

// --------------------------------------------------------------------- GEMM
// C[M,N] = A(MxK bf16) @ B(KxN, given as B^T row-major bf16) + bias[n]
// layout 0: C row-major MxN (bf16)
// layout 1: (B,H,S,HD)  : m=(b,s), n=(h,d)
// layout 2: (B,H,HD,S)  : transposed store (for V)
__global__ __launch_bounds__(128) void gemm_bf16_kernel(
        const unsigned short* __restrict__ A, const unsigned short* __restrict__ Bt,
        const float* __restrict__ bias, unsigned short* __restrict__ C,
        int M, int N, int Kd, int layout) {
    const int lane = threadIdx.x & 31;
    const int tile = blockIdx.x * 4 + (threadIdx.x >> 5);
    const int ntn  = N >> 6;
    const int tm   = tile / ntn;
    const int tn   = tile % ntn;
    if (tm * 16 >= M) return;

    v8f acc0 = {}, acc1 = {}, acc2 = {}, acc3 = {};
    for (int kk = 0; kk < Kd; kk += 32) {
        v16bf a = load_afrag(A, Kd, tm * 16, kk, lane);
        acc0 = wmma_bf16(a, load_bfrag(Bt, Kd, tn * 64 +  0, kk, lane), acc0);
        acc1 = wmma_bf16(a, load_bfrag(Bt, Kd, tn * 64 + 16, kk, lane), acc1);
        acc2 = wmma_bf16(a, load_bfrag(Bt, Kd, tn * 64 + 32, kk, lane), acc2);
        acc3 = wmma_bf16(a, load_bfrag(Bt, Kd, tn * 64 + 48, kk, lane), acc3);
    }
    v8f accs[4] = {acc0, acc1, acc2, acc3};
    const int mbase = tm * 16 + ((lane & 16) ? 8 : 0);
    #pragma unroll
    for (int g = 0; g < 4; ++g) {
        const int n  = tn * 64 + g * 16 + (lane & 15);
        const float bv = bias[n];
        #pragma unroll
        for (int e = 0; e < 8; ++e) {
            const int m = mbase + e;
            float val = accs[g][e] + bv;
            size_t idx;
            if (layout == 0) {
                idx = (size_t)m * N + n;
            } else {
                int b = m >> 9, s = m & 511, h = n >> 6, d = n & 63;
                if (layout == 1) idx = ((((size_t)b * 16 + h) * 512 + s) << 6) + d;
                else             idx = ((((size_t)b * 16 + h) * 64 + d) << 9) + s;
            }
            C[idx] = f2bf(val);
        }
    }
}

// ----------------------------------------------------------------- attention
// 1 wave = 16 query rows of one (b,h). Flash softmax over 32-key chunks.
// Disentangled bias computed on the fly via 48-wide banded WMMA GEMMs against
// PosK / PosQ, gathered along Toeplitz diagonals through per-wave LDS.
__global__ __launch_bounds__(128) void attn_kernel(
        const unsigned short* __restrict__ Qb, const unsigned short* __restrict__ Kb,
        const unsigned short* __restrict__ Vt, const unsigned short* __restrict__ posk,
        const unsigned short* __restrict__ posq, float* __restrict__ out) {
    __shared__ alignas(16) float          gc[4][16 * 48];   // c2p band tile
    __shared__ alignas(16) float          gp[4][32 * 48];   // p2c band tile
    __shared__ alignas(16) unsigned short pst[4][16 * 32];  // P staging (bf16)

    const int lane = threadIdx.x & 31;
    const int w    = threadIdx.x >> 5;
    const int bh   = blockIdx.x >> 3;
    const int b    = bh >> 4;
    const int h    = bh & 15;
    const int q0   = (blockIdx.x & 7) * 64 + w * 16;

    const unsigned short* Qh  = Qb + (size_t)bh * S_ * HD_;
    const unsigned short* Kh  = Kb + (size_t)bh * S_ * HD_;
    const unsigned short* Vh  = Vt + (size_t)bh * HD_ * S_;
    const unsigned short* PKh = posk + h * HD_;   // row stride 1024
    const unsigned short* PQh = posq + h * HD_;

    float*          gcw = gc[w];
    float*          gpw = gp[w];
    unsigned short* pw  = pst[w];

    const int hf = (lane & 16) ? 8 : 0;   // C-layout row offset for this half
    const int ln = lane & 15;             // C-layout column within tile

    v16bf a0 = load_afrag(Qh, HD_, q0, 0,  lane);   // Q d 0..31
    v16bf a1 = load_afrag(Qh, HD_, q0, 32, lane);   // Q d 32..63

    v8f o0 = {}, o1 = {}, o2 = {}, o3 = {};
    float mrow[8], lrow[8];
    #pragma unroll
    for (int r = 0; r < 8; ++r) { mrow[r] = -3.0e38f; lrow[r] = 0.0f; }

    for (int k0 = 0; k0 < S_; k0 += 32) {
        // ---- raw scores q.k (two 16-col halves) ----
        v8f slo = {}, shi = {};
        slo = wmma_bf16(a0, load_bfrag(Kh, HD_, k0,      0,  lane), slo);
        slo = wmma_bf16(a1, load_bfrag(Kh, HD_, k0,      32, lane), slo);
        shi = wmma_bf16(a0, load_bfrag(Kh, HD_, k0 + 16, 0,  lane), shi);
        shi = wmma_bf16(a1, load_bfrag(Kh, HD_, k0 + 16, 32, lane), shi);

        // ---- c2p band: G = Q_tile @ PosK[rb1 .. rb1+47]^T   (r = q-k+511) ----
        const int rb1 = q0 - k0 + 480;   // rb1 in [0, 976], band within [0,1023]
        #pragma unroll
        for (int t = 0; t < 3; ++t) {
            v8f g = {};
            g = wmma_bf16(a0, load_bfrag(PKh, 1024, rb1 + 16 * t, 0,  lane), g);
            g = wmma_bf16(a1, load_bfrag(PKh, 1024, rb1 + 16 * t, 32, lane), g);
            #pragma unroll
            for (int e = 0; e < 8; ++e) gcw[(e + hf) * 48 + 16 * t + ln] = g[e];
        }
        // ---- p2c band: G2 = K_chunk @ PosQ[rb2 .. rb2+47]^T (r' = k-q+511) ----
        const int rb2 = k0 - q0 + 496;
        #pragma unroll
        for (int tt = 0; tt < 2; ++tt) {
            v16bf ka0 = load_afrag(Kh, HD_, k0 + 16 * tt, 0,  lane);
            v16bf ka1 = load_afrag(Kh, HD_, k0 + 16 * tt, 32, lane);
            #pragma unroll
            for (int t = tt; t < tt + 2; ++t) {   // only the band tiles this sub-tile needs
                v8f g = {};
                g = wmma_bf16(ka0, load_bfrag(PQh, 1024, rb2 + 16 * t, 0,  lane), g);
                g = wmma_bf16(ka1, load_bfrag(PQh, 1024, rb2 + 16 * t, 32, lane), g);
                #pragma unroll
                for (int e = 0; e < 8; ++e)
                    gpw[(16 * tt + e + hf) * 48 + 16 * t + ln] = g[e];
            }
        }

        // ---- online softmax over this 16x32 score tile ----
        // (LDS from same wave is in-order -> no barrier needed)
        #pragma unroll
        for (int r = 0; r < 8; ++r) {
            const int qi  = r + hf;
            const int kl  = ln;
            const int kh2 = 16 + ln;
            float xlo = (slo[r] + gcw[qi * 48 + 31 + qi - kl]
                                + gpw[kl  * 48 + kl  - qi + 15]) * SCORE_SCALE;
            float xhi = (shi[r] + gcw[qi * 48 + 31 + qi - kh2]
                                + gpw[kh2 * 48 + kh2 - qi + 15]) * SCORE_SCALE;
            float vmax = fmaxf(xlo, xhi);
            vmax = fmaxf(vmax, __shfl_xor(vmax, 1, 32));
            vmax = fmaxf(vmax, __shfl_xor(vmax, 2, 32));
            vmax = fmaxf(vmax, __shfl_xor(vmax, 4, 32));
            vmax = fmaxf(vmax, __shfl_xor(vmax, 8, 32));
            float mnew = fmaxf(mrow[r], vmax);
            float corr = __expf(mrow[r] - mnew);
            float plo  = __expf(xlo - mnew);
            float phi  = __expf(xhi - mnew);
            float rs = plo + phi;
            rs += __shfl_xor(rs, 1, 32);
            rs += __shfl_xor(rs, 2, 32);
            rs += __shfl_xor(rs, 4, 32);
            rs += __shfl_xor(rs, 8, 32);
            lrow[r] = lrow[r] * corr + rs;
            mrow[r] = mnew;
            o0[r] *= corr; o1[r] *= corr; o2[r] *= corr; o3[r] *= corr;
            pw[qi * 32 + kl]  = f2bf(plo);
            pw[qi * 32 + kh2] = f2bf(phi);
        }

        // ---- O += P(16x32) @ V_chunk(32x64) ----
        v16bf pfrag = load_afrag(pw, 32, 0, 0, lane);   // re-layout C->A via LDS
        o0 = wmma_bf16(pfrag, load_bfrag(Vh, S_,  0, k0, lane), o0);
        o1 = wmma_bf16(pfrag, load_bfrag(Vh, S_, 16, k0, lane), o1);
        o2 = wmma_bf16(pfrag, load_bfrag(Vh, S_, 32, k0, lane), o2);
        o3 = wmma_bf16(pfrag, load_bfrag(Vh, S_, 48, k0, lane), o3);
    }

    // ---- epilogue: O / l, write (B, S, H*HD) fp32 ----
    v8f oarr[4] = {o0, o1, o2, o3};
    #pragma unroll
    for (int g = 0; g < 4; ++g) {
        const int d = g * 16 + ln;
        #pragma unroll
        for (int r = 0; r < 8; ++r) {
            const int q = q0 + r + hf;
            out[((size_t)b * 512 + q) * 1024 + h * 64 + d] = oarr[g][r] / lrow[r];
        }
    }
}

// ------------------------------------------------------------------- launch
extern "C" void kernel_launch(void* const* d_in, const int* in_sizes, int n_in,
                              void* d_out, int out_size, void* d_ws, size_t ws_size,
                              hipStream_t stream) {
    const float* x   = (const float*)d_in[0];
    const float* rel = (const float*)d_in[1];
    const float* Wq  = (const float*)d_in[2];
    const float* bq  = (const float*)d_in[3];
    const float* Wk  = (const float*)d_in[4];
    const float* bk  = (const float*)d_in[5];
    const float* Wv  = (const float*)d_in[6];
    const float* bv  = (const float*)d_in[7];
    float* out = (float*)d_out;

    // workspace carve-up (bf16 everywhere): total 28 MiB
    char* p = (char*)d_ws;
    unsigned short* xb    = (unsigned short*)p; p += (size_t)2048 * 1024 * 2; // x bf16
    unsigned short* relb  = (unsigned short*)p; p += (size_t)1024 * 1024 * 2; // rel bf16
    unsigned short* WqT   = (unsigned short*)p; p += (size_t)1024 * 1024 * 2;
    unsigned short* WkT   = (unsigned short*)p; p += (size_t)1024 * 1024 * 2;
    unsigned short* WvT   = (unsigned short*)p; p += (size_t)1024 * 1024 * 2;
    unsigned short* Qb    = (unsigned short*)p; p += (size_t)B_ * H_ * S_ * HD_ * 2;
    unsigned short* Kbuf  = (unsigned short*)p; p += (size_t)B_ * H_ * S_ * HD_ * 2;
    unsigned short* Vtb   = (unsigned short*)p; p += (size_t)B_ * H_ * HD_ * S_ * 2;
    unsigned short* poskb = (unsigned short*)p; p += (size_t)1024 * 1024 * 2;
    unsigned short* posqb = (unsigned short*)p; p += (size_t)1024 * 1024 * 2;

    // 1) precision conversion + weight transpose
    cvt_bf16_kernel<<<(2048 * 1024) / 256, 256, 0, stream>>>(x, xb, 2048 * 1024);
    cvt_bf16_kernel<<<(1024 * 1024) / 256, 256, 0, stream>>>(rel, relb, 1024 * 1024);
    transpose_cvt_kernel<<<(1024 * 1024) / 256, 256, 0, stream>>>(Wq, WqT);
    transpose_cvt_kernel<<<(1024 * 1024) / 256, 256, 0, stream>>>(Wk, WkT);
    transpose_cvt_kernel<<<(1024 * 1024) / 256, 256, 0, stream>>>(Wv, WvT);

    // 2) projections (WMMA GEMMs), V stored transposed, Pos* row-major
    gemm_bf16_kernel<<<512, 128, 0, stream>>>(xb,   WqT, bq, Qb,    2048, 1024, 1024, 1);
    gemm_bf16_kernel<<<512, 128, 0, stream>>>(xb,   WkT, bk, Kbuf,  2048, 1024, 1024, 1);
    gemm_bf16_kernel<<<512, 128, 0, stream>>>(xb,   WvT, bv, Vtb,   2048, 1024, 1024, 2);
    gemm_bf16_kernel<<<256, 128, 0, stream>>>(relb, WkT, bk, poskb, 1024, 1024, 1024, 0);
    gemm_bf16_kernel<<<256, 128, 0, stream>>>(relb, WqT, bq, posqb, 1024, 1024, 1024, 0);

    // 3) fused disentangled flash attention
    attn_kernel<<<B_ * H_ * (S_ / 64), 128, 0, stream>>>(Qb, Kbuf, Vtb, poskb, posqb, out);
}
